// LSTMHyperModel_29171417874452
// MI455X (gfx1250) — matrix operand order repaired and checked by reading, not compiled, not run on hardware
//
#include <hip/hip_runtime.h>
#include <hip/hip_bf16.h>
#include <math.h>
#include <stdint.h>

// ---------------- problem constants ----------------
#define B_   1024
#define T_   7
#define I_   5
#define H_   128
#define NV_  2100
#define KP_  2112          // NV padded up to multiple of 64
#define G4_  (4*H_)        // 512 gate rows

typedef __attribute__((ext_vector_type(16))) __bf16 v16bf;
typedef __attribute__((ext_vector_type(8)))  __bf16 v8bf;
typedef __attribute__((ext_vector_type(4)))  __bf16 v4bf;
typedef __attribute__((ext_vector_type(8)))  float  v8f;

// ---- CDNA5 async global->LDS copy (ASYNCcnt-tracked, no VGPR round trip) ----
// VGLOBAL async encoding: VDST = per-lane LDS byte address, VADDR = 64-bit
// global address (GV mode). Generic LDS pointers carry the wave-relative LDS
// offset in their low 32 bits.
__device__ __forceinline__ void async_copy_b128(void* lds_dst, const void* gsrc) {
  unsigned lds_addr = (unsigned)(uintptr_t)lds_dst;
  asm volatile("global_load_async_to_lds_b128 %0, %1, off"
               :: "v"(lds_addr), "v"(gsrc) : "memory");
}
__device__ __forceinline__ void wait_async0() {
  asm volatile("s_wait_asynccnt 0x0" ::: "memory");
}

// ================= pack news_vec f32 -> padded bf16 =================
__global__ void pack_a_kernel(const float* __restrict__ news, __bf16* __restrict__ out) {
  int idx = blockIdx.x * blockDim.x + threadIdx.x;   // over B_*KP_
  if (idx >= B_ * KP_) return;
  int b = idx / KP_, k = idx - b * KP_;
  float v = (k < NV_) ? news[(long)b * NV_ + k] : 0.0f;
  out[idx] = (__bf16)v;
}

// ================= hypernetwork GEMM (bf16 WMMA, f32 acc) =================
// C[m,n] = act( sum_k A[m,k] * (W1[n,k] (+W2[n,k])) + bias1[n] (+bias2[n]) )
#define BM 128
#define BN 128
#define BK 64
#define LDT 72   // padded LDS row stride (elems): 144B rows -> conflict-free

__global__ __launch_bounds__(256)
void hyper_gemm_kernel(const __bf16* __restrict__ A,     // [B_, KP_] bf16
                       const float*  __restrict__ W1,    // [N, NV_]
                       const float*  __restrict__ W2,    // nullable
                       const float*  __restrict__ bias1, // [N]
                       const float*  __restrict__ bias2, // nullable
                       int N, int applyTanh, int outIsBf16,
                       __bf16* __restrict__ outBf, float* __restrict__ outF)
{
  __shared__ __bf16 As[BM * LDT];
  __shared__ __bf16 Ws[BN * LDT];

  const int tid  = threadIdx.x;
  const int wave = tid >> 5;          // 0..7
  const int lane = tid & 31;
  const int wm   = wave >> 2;         // 0..1  -> 64 rows of M
  const int wn   = wave & 3;          // 0..3  -> 32 cols of N
  const int blockN = blockIdx.x * BN;
  const int blockM = blockIdx.y * BM;

  v8f acc[4][2];
  #pragma unroll
  for (int mt = 0; mt < 4; ++mt)
    #pragma unroll
    for (int nt = 0; nt < 2; ++nt)
      #pragma unroll
      for (int r = 0; r < 8; ++r) acc[mt][nt][r] = 0.0f;

  // cooperative tile loaders: 2 threads per row, 32 cols each
  const int lrow = tid >> 1;          // 0..127
  const int lcol = (tid & 1) * 32;    // 0 or 32

  for (int k0 = 0; k0 < KP_; k0 += BK) {
    // ---- A tile: async bf16 copy global->LDS (ASYNCcnt path) ----
    {
      const __bf16* src = A + (long)(blockM + lrow) * KP_ + k0 + lcol;
      __bf16* dst = &As[lrow * LDT + lcol];
      #pragma unroll
      for (int c = 0; c < 32; c += 8)
        async_copy_b128(dst + c, src + c);
    }
    // ---- W tile: f32 (+f32) -> bf16 ----
    {
      const long nrow = blockN + lrow;
      const float* s1 = W1 + nrow * NV_ + k0 + lcol;
      const float* s2 = W2 ? (W2 + nrow * NV_ + k0 + lcol) : nullptr;
      // prefetch next k-tile of W1 (global_prefetch_b8)
      if (k0 + BK < KP_) __builtin_prefetch(s1 + BK, 0, 1);
      if (k0 + BK <= NV_) {
        // fast path: full tile, vectorized (32 of 33 iterations)
        const float4* s1v = (const float4*)s1;
        const float4* s2v = (const float4*)s2;
        #pragma unroll
        for (int c4 = 0; c4 < 8; ++c4) {
          float4 v = s1v[c4];
          if (s2) {
            float4 w = s2v[c4];
            v.x += w.x; v.y += w.y; v.z += w.z; v.w += w.w;
          }
          v4bf o;
          o[0] = (__bf16)v.x; o[1] = (__bf16)v.y;
          o[2] = (__bf16)v.z; o[3] = (__bf16)v.w;
          *(v4bf*)&Ws[lrow * LDT + lcol + c4 * 4] = o;
        }
      } else {
        // ragged last K tile: per-element clamp
        #pragma unroll
        for (int c = 0; c < 32; ++c) {
          int k = k0 + lcol + c;
          float v = 0.0f;
          if (k < NV_) { v = s1[c]; if (s2) v += s2[c]; }
          Ws[lrow * LDT + lcol + c] = (__bf16)v;
        }
      }
    }
    wait_async0();
    __syncthreads();

    #pragma unroll
    for (int ks = 0; ks < 2; ++ks) {
      const int kb = ks * 32;
      v16bf a[4], bfr[2];
      // A fragments: lane L<16 holds K 0-7 & 16-23, L>=16 holds K 8-15 & 24-31
      const int aklo = kb + ((lane & 16) ? 8 : 0);
      #pragma unroll
      for (int mt = 0; mt < 4; ++mt) {
        int r = wm * 64 + mt * 16 + (lane & 15);
        v8bf lo = *(const v8bf*)&As[r * LDT + aklo];
        v8bf hi = *(const v8bf*)&As[r * LDT + aklo + 16];
        #pragma unroll
        for (int i = 0; i < 8; ++i) { a[mt][i] = lo[i]; a[mt][8 + i] = hi[i]; }
      }
      // B fragments: lane n = L%16; L<16 holds K 0-7 & 16-23, L>=16 holds K 8-15 & 24-31
      const int bklo = kb + ((lane & 16) ? 8 : 0);
      #pragma unroll
      for (int nt = 0; nt < 2; ++nt) {
        int n = wn * 32 + nt * 16 + (lane & 15);
        v8bf lo = *(const v8bf*)&Ws[n * LDT + bklo];
        v8bf hi = *(const v8bf*)&Ws[n * LDT + bklo + 16];
        #pragma unroll
        for (int i = 0; i < 8; ++i) { bfr[nt][i] = lo[i]; bfr[nt][8 + i] = hi[i]; }
      }
      #pragma unroll
      for (int mt = 0; mt < 4; ++mt)
        #pragma unroll
        for (int nt = 0; nt < 2; ++nt)
          acc[mt][nt] = __builtin_amdgcn_wmma_f32_16x16x32_bf16(
              false, a[mt], false, bfr[nt], (short)0, acc[mt][nt], false, false);
    }
    __syncthreads();
  }

  // ---- epilogue: +bias, optional tanh, store bf16 or f32 ----
  const int nloc  = lane & 15;
  const int mhalf = (lane >> 4) * 8;     // C/D layout: VGPR r -> row r / r+8
  #pragma unroll
  for (int nt = 0; nt < 2; ++nt) {
    int n = blockN + wn * 32 + nt * 16 + nloc;
    float bv = bias1[n];
    if (bias2) bv += bias2[n];
    #pragma unroll
    for (int mt = 0; mt < 4; ++mt) {
      #pragma unroll
      for (int r = 0; r < 8; ++r) {
        long m = blockM + wm * 64 + mt * 16 + mhalf + r;
        float v = acc[mt][nt][r] + bv;
        if (applyTanh) v = tanhf(v);
        if (outIsBf16) outBf[m * N + n] = (__bf16)v;
        else           outF [m * N + n] = v;
      }
    }
  }
}

// ================= per-sample LSTM, W_hh staged in LDS (128KB bf16) =================
__global__ __launch_bounds__(256)
void lstm_kernel(const __bf16* __restrict__ WhhAct,  // [B_, 512*128] bf16 (tanh'd)
                 const __bf16* __restrict__ WihAct,  // [B_, 512*5]   bf16 (tanh'd)
                 const float*  __restrict__ biasAct, // [B_, 512]
                 const float*  __restrict__ stock,   // [B_, T_, I_]
                 const float*  __restrict__ out_w,   // [H_]
                 const float*  __restrict__ out_b,   // [1]
                 float* __restrict__ out)            // [B_]
{
  extern __shared__ char smem[];
  __bf16* sWhh  = (__bf16*)smem;                                   // 512*128 bf16 = 128KB
  __bf16* sWih  = (__bf16*)(smem + (size_t)G4_ * H_ * 2);          // 512*8 bf16 (padded)
  float*  sBias = (float*)(smem + (size_t)G4_ * H_ * 2 + G4_ * 8 * 2);
  float*  sH = sBias + G4_;
  float*  sC = sH + H_;
  float*  sG = sC + H_;                                            // 512 gate pre-acts

  const int b   = blockIdx.x;
  const int tid = threadIdx.x;

  // stage W_hh (128KB) via async global->LDS B128 (ASYNCcnt path)
  {
    const char* src = (const char*)(WhhAct + (long)b * G4_ * H_);
    char* dst = (char*)sWhh;
    #pragma unroll 4
    for (int off = tid * 16; off < G4_ * H_ * 2; off += 256 * 16)
      async_copy_b128(dst + off, src + off);
  }
  for (int i = tid; i < G4_ * I_; i += 256) {
    int r = i / I_, c = i - r * I_;
    sWih[r * 8 + c] = WihAct[(long)b * G4_ * I_ + i];
  }
  for (int i = tid; i < G4_; i += 256) sBias[i] = biasAct[(long)b * G4_ + i];
  for (int i = tid; i < H_;  i += 256) { sH[i] = 0.0f; sC[i] = 0.0f; }
  wait_async0();
  __syncthreads();

  for (int t = 0; t < T_; ++t) {
    float x[I_];
    #pragma unroll
    for (int i = 0; i < I_; ++i) x[i] = stock[((long)b * T_ + t) * I_ + i];

    #pragma unroll
    for (int rr = 0; rr < 2; ++rr) {          // 2 gate rows per thread
      int r = tid * 2 + rr;
      float g = sBias[r];
      #pragma unroll
      for (int i = 0; i < I_; ++i) g += (float)sWih[r * 8 + i] * x[i];
      const __bf16* wrow = sWhh + r * H_;
      float acc = 0.0f;
      #pragma unroll 8
      for (int j = 0; j < H_; ++j) acc += (float)wrow[j] * sH[j];
      sG[r] = g + acc;
    }
    __syncthreads();

    if (tid < H_) {
      int j = tid;
      float ig = sG[j], fg = sG[j + 128], gg = sG[j + 256], og = sG[j + 384];
      float si = 1.0f / (1.0f + expf(-ig));
      float sf = 1.0f / (1.0f + expf(-fg));
      float so = 1.0f / (1.0f + expf(-og));
      float c  = sf * sC[j] + si * tanhf(gg);
      sC[j] = c;
      sH[j] = so * tanhf(c);
    }
    __syncthreads();
  }

  if (tid == 0) {
    float s = out_b[0];
    #pragma unroll 8
    for (int j = 0; j < H_; ++j) s += out_w[j] * sH[j];
    out[b] = s;
  }
}

// ================= launcher =================
extern "C" void kernel_launch(void* const* d_in, const int* in_sizes, int n_in,
                              void* d_out, int out_size, void* d_ws, size_t ws_size,
                              hipStream_t stream) {
  (void)in_sizes; (void)n_in; (void)out_size; (void)ws_size;
  const float* stock = (const float*)d_in[0];
  const float* news  = (const float*)d_in[1];
  const float* Wihw  = (const float*)d_in[2];
  const float* Wihb  = (const float*)d_in[3];
  const float* Whhw  = (const float*)d_in[4];
  const float* Whhb  = (const float*)d_in[5];
  const float* bihw  = (const float*)d_in[6];
  const float* bihb  = (const float*)d_in[7];
  const float* bhhw  = (const float*)d_in[8];
  const float* bhhb  = (const float*)d_in[9];
  const float* outw  = (const float*)d_in[10];
  const float* outb  = (const float*)d_in[11];
  float* out = (float*)d_out;

  // workspace layout
  char* ws = (char*)d_ws;
  size_t off = 0;
  __bf16* Abf    = (__bf16*)(ws + off); off += (size_t)B_ * KP_ * 2;          // 4.3 MB
  __bf16* WhhAct = (__bf16*)(ws + off); off += (size_t)B_ * G4_ * H_ * 2;     // 134 MB
  __bf16* WihAct = (__bf16*)(ws + off); off += (size_t)B_ * G4_ * I_ * 2;     // 5.2 MB
  float*  biasAct = (float*)(ws + off); off += (size_t)B_ * G4_ * 4;          // 2.1 MB

  // 1) pack A to bf16 (padded K)
  pack_a_kernel<<<(B_ * KP_ + 255) / 256, 256, 0, stream>>>(news, Abf);

  // 2) W_ih = tanh(news @ W_ih_w^T + b):  N = 2560
  hyper_gemm_kernel<<<dim3((4 * H_ * I_) / BN, B_ / BM), 256, 0, stream>>>(
      Abf, Wihw, nullptr, Wihb, nullptr, 4 * H_ * I_, 1, 1, WihAct, nullptr);

  // 3) W_hh = tanh(news @ W_hh_w^T + b):  N = 65536 (the 282-GFLOP GEMM)
  hyper_gemm_kernel<<<dim3((4 * H_ * H_) / BN, B_ / BM), 256, 0, stream>>>(
      Abf, Whhw, nullptr, Whhb, nullptr, 4 * H_ * H_, 1, 1, WhhAct, nullptr);

  // 4) bias = news @ (b_ih_w + b_hh_w)^T + (b_ih_b + b_hh_b):  N = 512, f32 out
  hyper_gemm_kernel<<<dim3(G4_ / BN, B_ / BM), 256, 0, stream>>>(
      Abf, bihw, bhhw, bihb, bhhb, G4_, 0, 0, nullptr, biasAct);

  // 5) recurrence: one block per sample, W_hh resident in LDS
  size_t shmem = (size_t)G4_ * H_ * 2      // sWhh 128KB
               + (size_t)G4_ * 8 * 2       // sWih padded
               + (size_t)(G4_ + H_ + H_ + G4_) * 4;   // bias, h, c, gates
  lstm_kernel<<<B_, 256, shmem, stream>>>(WhhAct, WihAct, biasAct, stock, outw, outb, out);
}
